// SupConLoss_72164040507507
// MI455X (gfx1250) — compile-verified, matrix operand order checked by reading, not run on hardware
//
#include <hip/hip_runtime.h>

// ---------------------------------------------------------------------------
// SupConLoss forward for MI455X (gfx1250, wave32, WMMA).
// N=4096 rows, D=128 feat dim, C=20 classes. Output: single f32 loss.
//
// Strategy: Gram matrix G = feats.featsT computed in 16x16 tiles with
// v_wmma_f32_16x16x32_f16 using a 2-term f16 split of the normalized f32
// features (hi*hi + hi*lo + lo*hi) for ~f32 accuracy. Both contrastive
// terms are streamed as online softmaxes over the tiles; G is never stored.
// Everything N^2-shaped downstream of the per-row scalars reduces to
// per-class (20) algebra done in a tiny finalize kernel.
// ---------------------------------------------------------------------------

#define NROWS 4096
#define DDIM 128
#define NCLS 20
#define NTILES (NROWS / 16)
#define WAVES 4

typedef __attribute__((ext_vector_type(16))) _Float16 v16h;
typedef __attribute__((ext_vector_type(8))) float v8f;

union FragU {
  v16h h;
  uint4 u[2];
};

// ---------------------------------------------------------------------------
// Kernel 1: L2-normalize rows (f32), emit hi/lo f16 split matrices.
// One 128-thread block per row.
// ---------------------------------------------------------------------------
__global__ __launch_bounds__(DDIM) void normalize_kernel(
    const float* __restrict__ X, _Float16* __restrict__ Fh,
    _Float16* __restrict__ Fl) {
  const int row = blockIdx.x;
  const int tid = threadIdx.x;
  float x = X[row * DDIM + tid];
  float ss = x * x;
#pragma unroll
  for (int off = 16; off >= 1; off >>= 1) ss += __shfl_xor(ss, off, 32);
  __shared__ float part[4];
  if ((tid & 31) == 0) part[tid >> 5] = ss;
  __syncthreads();
  const float tot = part[0] + part[1] + part[2] + part[3];
  const float nrm = fmaxf(sqrtf(tot), 1e-12f);
  const float xn = x / nrm;
  const _Float16 hi = (_Float16)xn;
  Fh[row * DDIM + tid] = hi;
  Fl[row * DDIM + tid] = (_Float16)(xn - (float)hi);
}

// ---------------------------------------------------------------------------
// Kernel 2: per-row cross entropy  ce[i] = logsumexp(p_i) - p_i[t_i]
// ---------------------------------------------------------------------------
__global__ __launch_bounds__(256) void ce_kernel(const float* __restrict__ P,
                                                 const int* __restrict__ T,
                                                 float* __restrict__ ce) {
  const int i = blockIdx.x * blockDim.x + threadIdx.x;
  if (i >= NROWS) return;
  float m = -3.4e38f;
#pragma unroll
  for (int c = 0; c < NCLS; ++c) m = fmaxf(m, P[i * NCLS + c]);
  float s = 0.f;
#pragma unroll
  for (int c = 0; c < NCLS; ++c) s += __expf(P[i * NCLS + c] - m);
  ce[i] = m + __logf(s) - P[i * NCLS + T[i]];
}

// ---------------------------------------------------------------------------
// Kernel 3: tiled Gram pass. One wave32 per 16-row strip.
// Outputs per row i:
//   rowmax[i] : max_j G[i,j]            (for nt_xent stabilization)
//   s1[i]     : sum_{j same,!=i} exp((G-rowmax)/0.1)
//   sG[i]     : sum_{j same,!=i} G[i,j]
//   r[i]      : sum_j exp(B[i,j] - max_j B[i,j]),  B = (G/0.05)*diff
// ---------------------------------------------------------------------------
__global__ __launch_bounds__(WAVES * 32) void gram_pass_kernel(
    const _Float16* __restrict__ Fh, const _Float16* __restrict__ Fl,
    const int* __restrict__ T, float* __restrict__ rowmax,
    float* __restrict__ s1out, float* __restrict__ sGout,
    float* __restrict__ rout) {
  const int wave = threadIdx.x >> 5;
  const int lane = threadIdx.x & 31;
  const int hf = lane >> 4;   // half-wave: selects K sub-block / C row group
  const int l16 = lane & 15;  // A row / B column within tile
  const int strip = blockIdx.x * WAVES + wave;

  // Preload A fragments (this wave's 16 rows x 128 K), hi and lo parts.
  // ISA 16-bit A layout: lane l (l<16) holds row l; element e of v16h maps to
  // K = (e>=8 ? 16 : 0) + hf*8 + (e&7)  ->  two contiguous 16B chunks.
  const _Float16* ArowH = Fh + (size_t)(strip * 16 + l16) * DDIM;
  const _Float16* ArowL = Fl + (size_t)(strip * 16 + l16) * DDIM;
  FragU aH[4], aL[4];
#pragma unroll
  for (int kc = 0; kc < 4; ++kc) {
    const _Float16* ph = ArowH + kc * 32 + hf * 8;
    const _Float16* pl = ArowL + kc * 32 + hf * 8;
    aH[kc].u[0] = *(const uint4*)(ph);
    aH[kc].u[1] = *(const uint4*)(ph + 16);
    aL[kc].u[0] = *(const uint4*)(pl);
    aL[kc].u[1] = *(const uint4*)(pl + 16);
  }

  int irow[8], trow[8];
#pragma unroll
  for (int v = 0; v < 8; ++v) {
    irow[v] = strip * 16 + hf * 8 + v;  // C/D layout: VGPR v -> M = v + 8*hf
    trow[v] = T[irow[v]];
  }

  float gmax[8], s1[8], sG[8], m2[8], s2[8];
#pragma unroll
  for (int v = 0; v < 8; ++v) {
    gmax[v] = -3.4e38f;
    s1[v] = 0.f;
    sG[v] = 0.f;
    m2[v] = -3.4e38f;
    s2[v] = 0.f;
  }

  const float invT = 10.f;   // 1/TEMP
  const float invT2 = 20.f;  // 1/TEMP2

  for (int jt = 0; jt < NTILES; ++jt) {
    const _Float16* BrowH = Fh + (size_t)(jt * 16 + l16) * DDIM;
    const _Float16* BrowL = Fl + (size_t)(jt * 16 + l16) * DDIM;
    v8f c = {0.f, 0.f, 0.f, 0.f, 0.f, 0.f, 0.f, 0.f};
#pragma unroll
    for (int kc = 0; kc < 4; ++kc) {
      FragU bH, bL;
      const _Float16* ph = BrowH + kc * 32 + hf * 8;
      const _Float16* pl = BrowL + kc * 32 + hf * 8;
      bH.u[0] = *(const uint4*)(ph);
      bH.u[1] = *(const uint4*)(ph + 16);
      bL.u[0] = *(const uint4*)(pl);
      bL.u[1] = *(const uint4*)(pl + 16);
      // Error-compensated split: G ~= hi*hi + hi*lo + lo*hi  (lo*lo ~ 2^-22)
      c = __builtin_amdgcn_wmma_f32_16x16x32_f16(false, aH[kc].h, false, bH.h,
                                                 (short)0, c, false, false);
      c = __builtin_amdgcn_wmma_f32_16x16x32_f16(false, aH[kc].h, false, bL.h,
                                                 (short)0, c, false, false);
      c = __builtin_amdgcn_wmma_f32_16x16x32_f16(false, aL[kc].h, false, bH.h,
                                                 (short)0, c, false, false);
    }
    const int jcol = jt * 16 + l16;
    const int tcol = T[jcol];
#pragma unroll
    for (int v = 0; v < 8; ++v) {
      const float g = c[v];
      const bool same = (trow[v] == tcol);
      // nt_xent: track full-row max of G; masked online exp-sum at 1/TEMP.
      const float mo = gmax[v];
      const float mn = fmaxf(mo, g);
      if (mn > mo) {
        s1[v] *= __expf((mo - mn) * invT);
        gmax[v] = mn;
      }
      if (same && (irow[v] != jcol)) {
        s1[v] += __expf((g - mn) * invT);
        sG[v] += g;
      }
      // nt_xent2: B entry is 0 for same-label, G/TEMP2 otherwise; r sums ALL j.
      const float bb = same ? 0.f : g * invT2;
      const float m2o = m2[v];
      const float m2n = fmaxf(m2o, bb);
      if (m2n > m2o) {
        s2[v] *= __expf(m2o - m2n);
        m2[v] = m2n;
      }
      s2[v] += __expf(bb - m2n);
    }
  }

  // Butterfly merge across the 16 lanes of each half-wave (online-softmax
  // combine for (gmax,s1) and (m2,s2); plain add for sG).
#pragma unroll
  for (int v = 0; v < 8; ++v) {
#pragma unroll
    for (int off = 1; off < 16; off <<= 1) {
      const float mo = __shfl_xor(gmax[v], off, 32);
      const float so = __shfl_xor(s1[v], off, 32);
      const float go = __shfl_xor(sG[v], off, 32);
      const float m2o = __shfl_xor(m2[v], off, 32);
      const float s2o = __shfl_xor(s2[v], off, 32);
      const float mn = fmaxf(gmax[v], mo);
      s1[v] = s1[v] * __expf((gmax[v] - mn) * invT) +
              so * __expf((mo - mn) * invT);
      gmax[v] = mn;
      sG[v] += go;
      const float m2n = fmaxf(m2[v], m2o);
      s2[v] = s2[v] * __expf(m2[v] - m2n) + s2o * __expf(m2o - m2n);
      m2[v] = m2n;
    }
  }

  if (l16 == 0) {
#pragma unroll
    for (int v = 0; v < 8; ++v) {
      const int i = irow[v];
      rowmax[i] = gmax[v];
      s1out[i] = s1[v];
      sGout[i] = sG[v];
      rout[i] = s2[v];
    }
  }
}

// ---------------------------------------------------------------------------
// Kernel 4: single-block deterministic finalize.
// ---------------------------------------------------------------------------
__device__ float block_reduce(float v, float* red, int tid) {
  red[tid] = v;
  __syncthreads();
  for (int s = 128; s > 0; s >>= 1) {
    if (tid < s) red[tid] += red[tid + s];
    __syncthreads();
  }
  const float r = red[0];
  __syncthreads();
  return r;
}

__global__ __launch_bounds__(256) void finalize_kernel(
    const int* __restrict__ T, const float* __restrict__ rowmax,
    const float* __restrict__ s1a, const float* __restrict__ sGa,
    const float* __restrict__ ra, const float* __restrict__ cea,
    float* __restrict__ out) {
  __shared__ float red[256];
  __shared__ int cnt[32];
  __shared__ float Rcls[32];
  const int tid = threadIdx.x;

  if (tid < 32) cnt[tid] = 0;
  __syncthreads();
  for (int i = tid; i < NROWS; i += 256) atomicAdd(&cnt[T[i]], 1);
  __syncthreads();

  // --- CE mean ---
  float acc = 0.f;
  for (int i = tid; i < NROWS; i += 256) acc += cea[i];
  const float ceSum = block_reduce(acc, red, tid);

  // --- supervised contrastive (nt_xent) mean ---
  acc = 0.f;
  for (int i = tid; i < NROWS; i += 256) {
    const int cE = cnt[T[i]] - 1;  // mask_sum (same-label, excl. diagonal)
    const float m1 = rowmax[i] * 10.f;  // rowmax of logits
    const float a1 = sGa[i] * 10.f - (float)cE * m1;
    const float row =
        (a1 - (float)cE * __logf(s1a[i] + 1e-12f)) / (float)(cE > 0 ? cE : 1);
    acc += row;
  }
  const float clSum = block_reduce(acc, red, tid);

  // --- per-class sums of r (fixed-order, deterministic) ---
  for (int c = 0; c < NCLS; ++c) {
    acc = 0.f;
    for (int i = tid; i < NROWS; i += 256) acc += (T[i] == c) ? ra[i] : 0.f;
    const float rc = block_reduce(acc, red, tid);
    if (tid == 0) Rcls[c] = rc;
  }
  __syncthreads();

  if (tid == 0) {
    // Label-only algebra for nt_xent_loss2:
    //   S_c      = (Rtot - Rcls[c]) + cnt[c]*N          (0 if no diff label)
    //   negsum_c = sum_c' cnt*(N-cnt) - cnt[c]*(N-cnt[c])
    float Rtot = 0.f, T1 = 0.f;
    for (int c = 0; c < NCLS; ++c) {
      Rtot += Rcls[c];
      const float nc = (float)cnt[c];
      T1 += nc * ((float)NROWS - nc);
    }
    bool all_zero = true;
    for (int c = 0; c < NCLS; ++c) {
      if (cnt[c] > 0) {
        const float nc = (float)cnt[c];
        if (T1 - nc * ((float)NROWS - nc) != 0.f) all_zero = false;
      }
    }
    float sum_cnt_logp = 0.f;
    for (int c = 0; c < NCLS; ++c) {
      if (cnt[c] == 0) continue;
      const float nsame = (float)cnt[c];
      const float cdiff = (float)NROWS - nsame;
      float S = (Rtot - Rcls[c]) + nsame * (float)NROWS;
      if (!(cdiff > 0.f)) S = 0.f;
      const float ns = T1 - nsame * cdiff;
      const float x = all_zero ? S : S / (ns == 0.f ? 1.f : ns);
      const float logp = -__logf(x + 1e-12f);
      sum_cnt_logp += nsame * logp;  // pos_logits_i == logp_{t_i}
    }
    const float triple = -sum_cnt_logp / (float)NROWS;
    const float ce = ceSum / (float)NROWS;
    const float cl = -clSum / (float)NROWS;
    // (1-ALPHA)*ce + ALPHA*cl + 0.5*ALPHA*triple, ALPHA=0.5
    out[0] = 0.5f * ce + 0.5f * cl + 0.25f * triple;
  }
}

// ---------------------------------------------------------------------------
extern "C" void kernel_launch(void* const* d_in, const int* in_sizes, int n_in,
                              void* d_out, int out_size, void* d_ws,
                              size_t ws_size, hipStream_t stream) {
  (void)in_sizes;
  (void)n_in;
  (void)out_size;
  (void)ws_size;
  const float* cls_feats = (const float*)d_in[0];
  const float* predicts = (const float*)d_in[1];
  const int* targets = (const int*)d_in[2];
  float* out = (float*)d_out;

  // Workspace layout:
  //   [0, 1MB)   : Fh  (NROWS*DDIM f16)
  //   [1MB, 2MB) : Fl  (NROWS*DDIM f16)
  //   then 5 x NROWS f32: rowmax, s1, sG, r, ce
  _Float16* Fh = (_Float16*)d_ws;
  _Float16* Fl = Fh + (size_t)NROWS * DDIM;
  float* base = (float*)(Fl + (size_t)NROWS * DDIM);
  float* rowmax = base + 0 * NROWS;
  float* s1 = base + 1 * NROWS;
  float* sG = base + 2 * NROWS;
  float* rr = base + 3 * NROWS;
  float* ce = base + 4 * NROWS;

  normalize_kernel<<<NROWS, DDIM, 0, stream>>>(cls_feats, Fh, Fl);
  ce_kernel<<<(NROWS + 255) / 256, 256, 0, stream>>>(predicts, targets, ce);
  gram_pass_kernel<<<NTILES / WAVES, WAVES * 32, 0, stream>>>(
      Fh, Fl, targets, rowmax, s1, sG, rr);
  finalize_kernel<<<1, 256, 0, stream>>>(targets, rowmax, s1, sG, rr, ce, out);
}
